// EAM_48034914239054
// MI455X (gfx1250) — matrix operand order repaired and checked by reading, not compiled, not run on hardware
//
#include <hip/hip_runtime.h>

// Fused 1x1-conv projections + attention (S = Q·Vᵀ, softmax, ctx = attn·K)
// One workgroup (8 wave32s) per n = b*64 + o.  Q/V/K tiles live in LDS as bf16,
// GEMMs use v_wmma_f32_16x16x32_bf16.  Output f32 (16,64,128,128).

typedef __attribute__((ext_vector_type(16))) __bf16 v16bf;
typedef __attribute__((ext_vector_type(8)))  float  v8f;

#define LDW 136   // padded LDS row stride (elements); 272B/row -> conflict-free 16-row frags

union BF16Frag { unsigned u[8]; v16bf v; };

__device__ __forceinline__ unsigned short f2bf(float f) {
  unsigned u = __builtin_bit_cast(unsigned, f);
  u += 0x7FFFu + ((u >> 16) & 1u);          // round-to-nearest-even
  return (unsigned short)(u >> 16);
}

__device__ __forceinline__ unsigned pack2(unsigned short lo, unsigned short hi) {
  return (unsigned)lo | ((unsigned)hi << 16);
}

// A fragment: 16x32 bf16, rows [row0, row0+16), K-chunk k0, from row-major LDS.
__device__ __forceinline__ v16bf ldsA(const unsigned short* base, int row0, int k0) {
  const int lane = threadIdx.x & 31;
  const int half = lane >> 4;
  const unsigned short* rp = base + (row0 + (lane & 15)) * LDW;
  BF16Frag f;
#pragma unroll
  for (int j = 0; j < 8; ++j) {
    const int kb = k0 + 2*j + ((j >= 4) ? 8 : 0) + 8*half;   // ISA 16-bit A layout
    f.u[j] = *(const unsigned*)(rp + kb);
  }
  return f.v;
}

// B fragment: 32x16 bf16, cols [n0, n0+16), K-chunk k0, from LDS holding B^T row-major.
__device__ __forceinline__ v16bf ldsB(const unsigned short* base, int n0, int k0) {
  const int lane = threadIdx.x & 31;
  const int half = lane >> 4;
  const unsigned short* rp = base + (n0 + (lane & 15)) * LDW;
  BF16Frag f;
#pragma unroll
  for (int j = 0; j < 8; ++j) {
    const int kb = k0 + 2*j + ((j >= 4) ? 8 : 0) + 8*half;   // ISA 16-bit B layout
    f.u[j] = *(const unsigned*)(rp + kb);
  }
  return f.v;
}

__global__ __launch_bounds__(256)
void EAM_attn_fused(const float* __restrict__ x, const float* __restrict__ e,
                    const float* __restrict__ W1, const float* __restrict__ b1,
                    const float* __restrict__ W2, const float* __restrict__ b2,
                    const float* __restrict__ W3, const float* __restrict__ b3,
                    float* __restrict__ out) {
  __shared__ __align__(16) unsigned short Qs [128 * LDW]; // Q[h][w]; reused as attn[h][g]
  __shared__ __align__(16) unsigned short Vs [128 * LDW]; // V[g][w]  (= B^T for GEMM1)
  __shared__ __align__(16) unsigned short KsT[128 * LDW]; // K^T[w][g] (= B^T for GEMM2)
  __shared__ float wgt[144];                               // W1 row | W2 row | W3 row

  const int n   = blockIdx.x;
  const int b   = n >> 6;
  const int o   = n & 63;
  const int tid = threadIdx.x;

  if (tid < 16)       wgt[tid]            = W1[o * 16 + tid];
  if (tid < 64)       wgt[16 + tid]       = W2[o * 64 + tid];
  else if (tid < 128) wgt[80 + tid - 64]  = W3[o * 64 + tid - 64];
  __syncthreads();

  const float bq = b1[o], bv = b2[o], bk = b3[o];
  const float* ep = e + (size_t)b * 16 * 16384;
  const float* xp = x + (size_t)b * 64 * 16384;

  // ---- Projections: Q = e·W1ᵀ+b1, V = x·W2ᵀ+b2, K = x·W3ᵀ+b3 (bf16 into LDS) ----
  // Each thread owns 4 CONTIGUOUS pixels -> float4 (global_load_b128) per channel,
  // fully coalesced (512B per wave per instruction).
  for (int blk = 0; blk < 16384; blk += 1024) {
    const int p0 = blk + (tid << 2);          // 4-aligned, same row h for all 4 pixels
    float q[4], vv[4], kk[4];
#pragma unroll
    for (int j = 0; j < 4; ++j) { q[j] = bq; vv[j] = bv; kk[j] = bk; }
#pragma unroll
    for (int c = 0; c < 16; ++c) {
      const float  wq  = wgt[c];
      const float4 ev4 = *(const float4*)(ep + c * 16384 + p0);
      q[0] = fmaf(wq, ev4.x, q[0]);
      q[1] = fmaf(wq, ev4.y, q[1]);
      q[2] = fmaf(wq, ev4.z, q[2]);
      q[3] = fmaf(wq, ev4.w, q[3]);
    }
#pragma unroll 4
    for (int c = 0; c < 64; ++c) {
      const float  w2  = wgt[16 + c], w3 = wgt[80 + c];
      const float4 xv4 = *(const float4*)(xp + c * 16384 + p0);
      vv[0] = fmaf(w2, xv4.x, vv[0]);  kk[0] = fmaf(w3, xv4.x, kk[0]);
      vv[1] = fmaf(w2, xv4.y, vv[1]);  kk[1] = fmaf(w3, xv4.y, kk[1]);
      vv[2] = fmaf(w2, xv4.z, vv[2]);  kk[2] = fmaf(w3, xv4.z, kk[2]);
      vv[3] = fmaf(w2, xv4.w, vv[3]);  kk[3] = fmaf(w3, xv4.w, kk[3]);
    }
    const int h  = p0 >> 7;
    const int w0 = p0 & 127;
    // Q, V: 4 consecutive bf16 in one row -> single 8B LDS store each.
    uint2 qp, vp;
    qp.x = pack2(f2bf(q[0]),  f2bf(q[1]));
    qp.y = pack2(f2bf(q[2]),  f2bf(q[3]));
    vp.x = pack2(f2bf(vv[0]), f2bf(vv[1]));
    vp.y = pack2(f2bf(vv[2]), f2bf(vv[3]));
    *(uint2*)(&Qs[h * LDW + w0]) = qp;        // 8B-aligned: h*136 + w0 ≡ 0 (mod 4) elems
    *(uint2*)(&Vs[h * LDW + w0]) = vp;
#pragma unroll
    for (int j = 0; j < 4; ++j)               // K stored transposed: [w][h]
      KsT[(w0 + j) * LDW + h] = f2bf(kk[j]);
  }
  __syncthreads();

  const int wv   = tid >> 5;
  const int m0   = wv * 16;          // this wave's 16-row strip
  const int lane = tid & 31;
  const int half = lane >> 4;

  // ---- GEMM1: S[h][g] = sum_w Q[h][w] * V[g][w] ----
  v8f acc[8];
#pragma unroll
  for (int t = 0; t < 8; ++t) acc[t] = v8f{0.f, 0.f, 0.f, 0.f, 0.f, 0.f, 0.f, 0.f};
#pragma unroll
  for (int k0 = 0; k0 < 128; k0 += 32) {
    const v16bf a = ldsA(Qs, m0, k0);
#pragma unroll
    for (int t = 0; t < 8; ++t) {
      const v16bf bb = ldsB(Vs, t * 16, k0);
      acc[t] = __builtin_amdgcn_wmma_f32_16x16x32_bf16(false, a, false, bb,
                                                       (short)0, acc[t], false, false);
    }
  }

  // ---- Row softmax over g; write attn (bf16) over Qs' strip (same wave, in-order LDS) ----
#pragma unroll
  for (int i = 0; i < 8; ++i) {
    float m = -3.4e38f;
#pragma unroll
    for (int t = 0; t < 8; ++t) m = fmaxf(m, acc[t][i]);
    for (int off = 8; off >= 1; off >>= 1) m = fmaxf(m, __shfl_xor(m, off, 32));
    float s = 0.f;
#pragma unroll
    for (int t = 0; t < 8; ++t) { const float ev = __expf(acc[t][i] - m); acc[t][i] = ev; s += ev; }
    for (int off = 8; off >= 1; off >>= 1) s += __shfl_xor(s, off, 32);
    const float inv = 1.0f / s;
    const int row = m0 + i + 8 * half;   // C/D layout: lanes 16-31 carry rows +8
#pragma unroll
    for (int t = 0; t < 8; ++t)
      Qs[row * LDW + t * 16 + (lane & 15)] = f2bf(acc[t][i] * inv);
  }

  // ---- GEMM2: ctx[h][w] = sum_g attn[h][g] * K[g][w] ----
#pragma unroll
  for (int t = 0; t < 8; ++t) acc[t] = v8f{0.f, 0.f, 0.f, 0.f, 0.f, 0.f, 0.f, 0.f};
#pragma unroll
  for (int k0 = 0; k0 < 128; k0 += 32) {
    const v16bf a = ldsA(Qs, m0, k0);          // attn strip
#pragma unroll
    for (int t = 0; t < 8; ++t) {
      const v16bf bb = ldsB(KsT, t * 16, k0);  // B^T = K^T row-major
      acc[t] = __builtin_amdgcn_wmma_f32_16x16x32_bf16(false, a, false, bb,
                                                       (short)0, acc[t], false, false);
    }
  }

  // ---- Store ctx (f32) ----
  float* op = out + (size_t)n * 16384;
#pragma unroll
  for (int i = 0; i < 8; ++i) {
    const int row = m0 + i + 8 * half;
#pragma unroll
    for (int t = 0; t < 8; ++t)
      op[row * 128 + t * 16 + (lane & 15)] = acc[t][i];
  }
}

extern "C" void kernel_launch(void* const* d_in, const int* in_sizes, int n_in,
                              void* d_out, int out_size, void* d_ws, size_t ws_size,
                              hipStream_t stream) {
  (void)in_sizes; (void)n_in; (void)out_size; (void)d_ws; (void)ws_size;
  const float* x  = (const float*)d_in[0];
  const float* e  = (const float*)d_in[1];
  const float* W1 = (const float*)d_in[2];
  const float* b1 = (const float*)d_in[3];
  const float* W2 = (const float*)d_in[4];
  const float* b2 = (const float*)d_in[5];
  const float* W3 = (const float*)d_in[6];
  const float* b3 = (const float*)d_in[7];
  EAM_attn_fused<<<dim3(1024), dim3(256), 0, stream>>>(
      x, e, W1, b1, W2, b2, W3, b3, (float*)d_out);
}
